// MultiHeadAttention_11012296147544
// MI455X (gfx1250) — matrix operand order repaired and checked by reading, not compiled, not run on hardware
//
#include <hip/hip_runtime.h>
#include <hip/hip_bf16.h>

typedef _Float16 v16h __attribute__((ext_vector_type(16)));
typedef _Float16 v8h  __attribute__((ext_vector_type(8)));
typedef float    v8f  __attribute__((ext_vector_type(8)));
typedef unsigned int v4u __attribute__((ext_vector_type(4)));
typedef int      v8i  __attribute__((ext_vector_type(8)));
typedef int      v4i  __attribute__((ext_vector_type(4)));

constexpr int kB  = 4;
constexpr int kS  = 2048;
constexpr int kD  = 1024;
constexpr int kH  = 16;
constexpr int kHD = 64;
constexpr int kM  = kB * kS;   // 8192 rows for the projection GEMMs

__device__ __forceinline__ v16h mk16(v8h lo, v8h hi) {
  v16h r;
#pragma unroll
  for (int i = 0; i < 8; ++i) { r[i] = lo[i]; r[i + 8] = hi[i]; }
  return r;
}

__device__ __forceinline__ v8f wmma_f16(v16h a, v16h b, v8f c) {
  // D = A(16x32 f16) * B(32x16 f16) + C(16x16 f32)
  return __builtin_amdgcn_wmma_f32_16x16x32_f16(false, a, false, b, (short)0, c,
                                                false, false);
}

// ---------------------------------------------------------------------------
// TDM: 2D tile (f16 elements) global -> LDS, with optional LDS row padding.
// Descriptor per CDNA5 ISA ch.8: group0 {count,lds,global,type=2},
// group1 {data_size=2B, pad, dims, tile dims, dim0 stride}. Groups 2/3 zero
// (2D tile: tile_dim2 = 0). Issued by one wave; tracked by TENSORcnt.
// ---------------------------------------------------------------------------
__device__ __forceinline__ void tdm_load_2d_f16(
    unsigned int lds_byte, unsigned long long gaddr_byte,
    unsigned int tensor_d0, unsigned int tensor_d1,
    unsigned int tile_d0, unsigned int tile_d1,
    unsigned long long d0_stride_elems,
    unsigned int pad_interval_code, unsigned int pad_amount_code) {
  v4u g0;
  g0[0] = 1u;  // count=1 valid descriptor; gather/iterate/restore off
  g0[1] = lds_byte;
  g0[2] = (unsigned int)(gaddr_byte & 0xffffffffu);
  g0[3] = (unsigned int)((gaddr_byte >> 32) & 0x01ffffffu) | (2u << 30);  // type=2
  v8i g1;
  const unsigned int pad_en = (pad_amount_code || pad_interval_code) ? 1u : 0u;
  g1[0] = (int)((1u << 16)                 // data_size = 2 bytes
              | (pad_en << 20)
              | (pad_interval_code << 22)
              | (pad_amount_code << 25));
  g1[1] = (int)((tensor_d0 & 0xffffu) << 16);
  g1[2] = (int)(((tensor_d0 >> 16) & 0xffffu) | ((tensor_d1 & 0xffffu) << 16));
  g1[3] = (int)(((tensor_d1 >> 16) & 0xffffu) | ((tile_d0 & 0xffffu) << 16));
  g1[4] = (int)(tile_d1 & 0xffffu);        // tile_dim1; tile_dim2 = 0
  g1[5] = (int)(unsigned int)(d0_stride_elems & 0xffffffffu);
  g1[6] = (int)(unsigned int)((d0_stride_elems >> 32) & 0xffffu);  // dim1_stride=0
  g1[7] = 0;
  const v4i z4 = {0, 0, 0, 0};
#if defined(__clang_major__) && __clang_major__ >= 23
  const v8i z8 = {0, 0, 0, 0, 0, 0, 0, 0};
  __builtin_amdgcn_tensor_load_to_lds(g0, g1, z4, z4, z8, 0);
#else
  __builtin_amdgcn_tensor_load_to_lds(g0, g1, z4, z4, 0);
#endif
}

// ---------------------------------------------------------------------------
// GEMM: Out[m, n] = X[m, :] . W[n, :] + bias[n]   (torch Linear semantics)
// fp32 inputs converted to f16, output f16:
//   VMODE=0: head-major [B,H,S,HD]   (Q, K)
//   VMODE=1: head-major transposed [B,H,HD,S]  (V, so attention can TDM-load
//            a [d][t] tile directly; TDM cannot transpose)
// block tile: 128(M) x 64(N), 8 waves -> each wave 32x32 (2x2 WMMA tiles)
// ---------------------------------------------------------------------------
template <int VMODE>
__global__ __launch_bounds__(256) void gemm_qkv_kernel(
    const float* __restrict__ X, const float* __restrict__ W,
    const float* __restrict__ bias, _Float16* __restrict__ Out) {
  constexpr int BM = 128, BN = 64, BK = 32, LDT = BK + 8;
  __shared__ __align__(16) _Float16 As[BM][LDT];
  __shared__ __align__(16) _Float16 Bs[BN][LDT];

  const int tid  = threadIdx.x;
  const int wave = tid >> 5;
  const int lane = tid & 31;
  const int l15  = lane & 15;
  const int lh   = lane >> 4;
  const int m0 = blockIdx.x * BM;
  const int n0 = blockIdx.y * BN;
  const int wm = (wave & 3) * 32;   // wave row offset in block tile
  const int wn = (wave >> 2) * 32;  // wave col offset in block tile

  v8f acc[2][2];
#pragma unroll
  for (int i = 0; i < 2; ++i)
#pragma unroll
    for (int j = 0; j < 2; ++j)
#pragma unroll
      for (int e = 0; e < 8; ++e) acc[i][j][e] = 0.0f;

  const int ar = (tid * 16) >> 5, ac = (tid * 16) & 31;  // 128x32 fill, 16/thr
  const int br = (tid * 8) >> 5,  bc = (tid * 8) & 31;   // 64x32 fill, 8/thr

  for (int k0 = 0; k0 < kD; k0 += BK) {
    __syncthreads();
    const float* ax = X + (size_t)(m0 + ar) * kD + k0 + ac;
#pragma unroll
    for (int i = 0; i < 16; ++i) As[ar][ac + i] = (_Float16)ax[i];
    const float* bx = W + (size_t)(n0 + br) * kD + k0 + bc;
#pragma unroll
    for (int i = 0; i < 8; ++i) Bs[br][bc + i] = (_Float16)bx[i];
    if (k0 + BK < kD) {
      __builtin_prefetch(ax + BK, 0, 0);
      __builtin_prefetch(bx + BK, 0, 0);
    }
    __syncthreads();

    v16h af[2], bf[2];
#pragma unroll
    for (int i = 0; i < 2; ++i) {
      const int row = wm + i * 16 + l15;
      af[i] = mk16(*(const v8h*)&As[row][lh * 8],
                   *(const v8h*)&As[row][16 + lh * 8]);
    }
#pragma unroll
    for (int j = 0; j < 2; ++j) {
      const int col = wn + j * 16 + l15;
      bf[j] = mk16(*(const v8h*)&Bs[col][lh * 8],
                   *(const v8h*)&Bs[col][16 + lh * 8]);
    }
#pragma unroll
    for (int i = 0; i < 2; ++i)
#pragma unroll
      for (int j = 0; j < 2; ++j) acc[i][j] = wmma_f16(af[i], bf[j], acc[i][j]);
  }

#pragma unroll
  for (int i = 0; i < 2; ++i)
#pragma unroll
    for (int j = 0; j < 2; ++j) {
      const int n   = n0 + wn + j * 16 + l15;
      const float bb = bias[n];
      const int h   = n >> 6;          // n / HD
      const int hd  = n & (kHD - 1);
#pragma unroll
      for (int e = 0; e < 8; ++e) {
        const int m = m0 + wm + i * 16 + e + lh * 8;
        const int b = m >> 11;         // m / S
        const int s = m & (kS - 1);
        const _Float16 val = (_Float16)(acc[i][j][e] + bb);
        if (VMODE)
          Out[(((size_t)b * kH + h) * kHD + hd) * kS + s] = val;
        else
          Out[(((size_t)b * kH + h) * kS + s) * kHD + hd] = val;
      }
    }
}

// ---------------------------------------------------------------------------
// Output projection: f16 A, fp32 Out (row-major [M, D]) + bias
// ---------------------------------------------------------------------------
__global__ __launch_bounds__(256) void gemm_out_kernel(
    const _Float16* __restrict__ X, const float* __restrict__ W,
    const float* __restrict__ bias, float* __restrict__ Out) {
  constexpr int BM = 128, BN = 64, BK = 32, LDT = BK + 8;
  __shared__ __align__(16) _Float16 As[BM][LDT];
  __shared__ __align__(16) _Float16 Bs[BN][LDT];

  const int tid  = threadIdx.x;
  const int wave = tid >> 5;
  const int lane = tid & 31;
  const int l15  = lane & 15;
  const int lh   = lane >> 4;
  const int m0 = blockIdx.x * BM;
  const int n0 = blockIdx.y * BN;
  const int wm = (wave & 3) * 32;
  const int wn = (wave >> 2) * 32;

  v8f acc[2][2];
#pragma unroll
  for (int i = 0; i < 2; ++i)
#pragma unroll
    for (int j = 0; j < 2; ++j)
#pragma unroll
      for (int e = 0; e < 8; ++e) acc[i][j][e] = 0.0f;

  const int ar = (tid * 16) >> 5, ac = (tid * 16) & 31;
  const int br = (tid * 8) >> 5,  bc = (tid * 8) & 31;

  for (int k0 = 0; k0 < kD; k0 += BK) {
    __syncthreads();
    const _Float16* ax = X + (size_t)(m0 + ar) * kD + k0 + ac;
    *(v8h*)&As[ar][ac]     = *(const v8h*)ax;
    *(v8h*)&As[ar][ac + 8] = *(const v8h*)(ax + 8);
    const float* bx = W + (size_t)(n0 + br) * kD + k0 + bc;
#pragma unroll
    for (int i = 0; i < 8; ++i) Bs[br][bc + i] = (_Float16)bx[i];
    if (k0 + BK < kD) {
      __builtin_prefetch(ax + BK, 0, 0);
      __builtin_prefetch(bx + BK, 0, 0);
    }
    __syncthreads();

    v16h af[2], bf[2];
#pragma unroll
    for (int i = 0; i < 2; ++i) {
      const int row = wm + i * 16 + l15;
      af[i] = mk16(*(const v8h*)&As[row][lh * 8],
                   *(const v8h*)&As[row][16 + lh * 8]);
    }
#pragma unroll
    for (int j = 0; j < 2; ++j) {
      const int col = wn + j * 16 + l15;
      bf[j] = mk16(*(const v8h*)&Bs[col][lh * 8],
                   *(const v8h*)&Bs[col][16 + lh * 8]);
    }
#pragma unroll
    for (int i = 0; i < 2; ++i)
#pragma unroll
      for (int j = 0; j < 2; ++j) acc[i][j] = wmma_f16(af[i], bf[j], acc[i][j]);
  }

#pragma unroll
  for (int i = 0; i < 2; ++i)
#pragma unroll
    for (int j = 0; j < 2; ++j) {
      const int n = n0 + wn + j * 16 + l15;
      const float bb = bias[n];
#pragma unroll
      for (int e = 0; e < 8; ++e) {
        const int m = m0 + wm + i * 16 + e + lh * 8;
        Out[(size_t)m * kD + n] = acc[i][j][e] + bb;
      }
    }
}

// ---------------------------------------------------------------------------
// Flash-style causal attention: one block per (b, h, 64-row q tile).
// 4 waves, each wave owns 16 q rows; online softmax; only kt <= qt tiles.
// K [t][d] and pre-transposed V [d][t] tiles are staged into padded LDS by
// the Tensor Data Mover (wave 0 issues, s_wait_tensorcnt, block barrier).
// Writes merged-head f16 output [B, S, D].
// ---------------------------------------------------------------------------
__global__ __launch_bounds__(128) void attn_kernel(
    const _Float16* __restrict__ Q, const _Float16* __restrict__ Kh,
    const _Float16* __restrict__ Vt, _Float16* __restrict__ O) {
  constexpr int QT = 64, KT = 64, LDK = kHD + 8, LDV = KT + 8;
  // Row stride 72 halves = 144B: TDM pad_interval=32 DWORDs (code 4),
  // pad_amount=4 DWORDs (code 3) reproduces this padding.
  __shared__ __align__(16) _Float16 kls[KT][LDK];     // [t][d]
  __shared__ __align__(16) _Float16 vls[kHD][LDV];    // [d][t]
  __shared__ __align__(16) _Float16 pls[4][16][KT];   // per-wave P staging

  const int tid  = threadIdx.x;
  const int wave = tid >> 5;
  const int lane = tid & 31;
  const int l15  = lane & 15;
  const int lh   = lane >> 4;
  const int bh = blockIdx.y;            // b*H + h
  const int b  = bh >> 4;
  const int h  = bh & (kH - 1);
  const int q0 = blockIdx.x * QT;
  const size_t hb = (size_t)bh * kS * kHD;

  // Q fragments (A layout): rows q0 + wave*16 + (lane&15)
  const _Float16* qp = Q + hb + (size_t)(q0 + wave * 16 + l15) * kHD;
  v16h qa[2];
#pragma unroll
  for (int ks = 0; ks < 2; ++ks)
    qa[ks] = mk16(*(const v8h*)(qp + ks * 32 + lh * 8),
                  *(const v8h*)(qp + ks * 32 + 16 + lh * 8));

  v8f oacc[4];
#pragma unroll
  for (int dt = 0; dt < 4; ++dt)
#pragma unroll
    for (int e = 0; e < 8; ++e) oacc[dt][e] = 0.0f;
  float mrow[8], lrow[8];
#pragma unroll
  for (int e = 0; e < 8; ++e) { mrow[e] = -3.0e38f; lrow[e] = 0.0f; }

  const unsigned int kls_lds = (unsigned int)(unsigned long long)&kls[0][0];
  const unsigned int vls_lds = (unsigned int)(unsigned long long)&vls[0][0];

  const int nkt = blockIdx.x + 1;  // causal: only key tiles <= q tile
  for (int kt = 0; kt < nkt; ++kt) {
    const int kb = kt * KT;
    __syncthreads();  // previous iteration finished reading kls/vls
    if (wave == 0) {
      // K tile: rows t = kb..kb+63 (stride HD), cols d = 0..63
      tdm_load_2d_f16(kls_lds,
                      (unsigned long long)(const void*)(Kh + hb) +
                          (unsigned long long)kb * kHD * 2ull,
                      /*tensor_d0=*/kHD, /*tensor_d1=*/kS,
                      /*tile_d0=*/kHD, /*tile_d1=*/KT,
                      /*d0_stride=*/kHD, /*pad_int=*/4u, /*pad_amt=*/3u);
      // V tile (pre-transposed layout): rows d = 0..63 (stride S), cols t
      tdm_load_2d_f16(vls_lds,
                      (unsigned long long)(const void*)(Vt + (size_t)bh * kHD * kS) +
                          (unsigned long long)kb * 2ull,
                      /*tensor_d0=*/kS, /*tensor_d1=*/kHD,
                      /*tile_d0=*/KT, /*tile_d1=*/kHD,
                      /*d0_stride=*/kS, /*pad_int=*/4u, /*pad_amt=*/3u);
      __builtin_amdgcn_s_wait_tensorcnt(0);
    }
    __syncthreads();  // publish TDM-written LDS to all waves

    // S = Q K^T  (16 x 64 per wave, contraction over HD=64)
    v8f sacc[4];
#pragma unroll
    for (int nt = 0; nt < 4; ++nt)
#pragma unroll
      for (int e = 0; e < 8; ++e) sacc[nt][e] = 0.0f;
#pragma unroll
    for (int nt = 0; nt < 4; ++nt) {
      const int kr = nt * 16 + l15;
#pragma unroll
      for (int ks = 0; ks < 2; ++ks) {
        v16h bf = mk16(*(const v8h*)&kls[kr][ks * 32 + lh * 8],
                       *(const v8h*)&kls[kr][ks * 32 + 16 + lh * 8]);
        sacc[nt] = wmma_f16(qa[ks], bf, sacc[nt]);
      }
    }

    // scale + causal mask (only the diagonal tile needs masking)
    const bool diag = (kt == blockIdx.x);
#pragma unroll
    for (int nt = 0; nt < 4; ++nt)
#pragma unroll
      for (int e = 0; e < 8; ++e) {
        float s = sacc[nt][e] * 0.125f;  // 1/sqrt(HD)
        if (diag) {
          const int col = kb + nt * 16 + l15;
          const int row = q0 + wave * 16 + e + lh * 8;
          if (col > row) s = -1.0e30f;
        }
        sacc[nt][e] = s;
      }

    // online softmax: per-row max/sum via xor-shuffles within 16-lane halves
#pragma unroll
    for (int e = 0; e < 8; ++e) {
      float mx = fmaxf(fmaxf(sacc[0][e], sacc[1][e]),
                       fmaxf(sacc[2][e], sacc[3][e]));
#pragma unroll
      for (int off = 1; off < 16; off <<= 1)
        mx = fmaxf(mx, __shfl_xor(mx, off, 32));
      const float mn = fmaxf(mrow[e], mx);
      const float f  = exp2f((mrow[e] - mn) * 1.44269504f);
      mrow[e] = mn;
      lrow[e] *= f;
#pragma unroll
      for (int dt = 0; dt < 4; ++dt) oacc[dt][e] *= f;
      float rs = 0.0f;
#pragma unroll
      for (int nt = 0; nt < 4; ++nt) {
        const float p = exp2f((sacc[nt][e] - mn) * 1.44269504f);
        sacc[nt][e] = p;
        rs += p;
      }
#pragma unroll
      for (int off = 1; off < 16; off <<= 1) rs += __shfl_xor(rs, off, 32);
      lrow[e] += rs;
    }

    // P: C-layout -> LDS -> A-layout fragments
#pragma unroll
    for (int nt = 0; nt < 4; ++nt)
#pragma unroll
      for (int e = 0; e < 8; ++e)
        pls[wave][e + lh * 8][nt * 16 + l15] = (_Float16)sacc[nt][e];
    __syncthreads();

    v16h pa[2];
#pragma unroll
    for (int ks = 0; ks < 2; ++ks)
      pa[ks] = mk16(*(const v8h*)&pls[wave][l15][ks * 32 + lh * 8],
                    *(const v8h*)&pls[wave][l15][ks * 32 + 16 + lh * 8]);

    // O += P V  (contraction over t)
#pragma unroll
    for (int dt = 0; dt < 4; ++dt) {
      const int dr = dt * 16 + l15;
#pragma unroll
      for (int ks = 0; ks < 2; ++ks) {
        v16h vf = mk16(*(const v8h*)&vls[dr][ks * 32 + lh * 8],
                       *(const v8h*)&vls[dr][ks * 32 + 16 + lh * 8]);
        oacc[dt] = wmma_f16(pa[ks], vf, oacc[dt]);
      }
    }
  }

  // normalize + write merged heads [B, S, D]
#pragma unroll
  for (int dt = 0; dt < 4; ++dt)
#pragma unroll
    for (int e = 0; e < 8; ++e) {
      const int d   = dt * 16 + l15;
      const int row = q0 + wave * 16 + e + lh * 8;
      const float o = oacc[dt][e] / lrow[e];
      O[((size_t)b * kS + row) * kD + h * kHD + d] = (_Float16)o;
    }
}

// ---------------------------------------------------------------------------
extern "C" void kernel_launch(void* const* d_in, const int* in_sizes, int n_in,
                              void* d_out, int out_size, void* d_ws,
                              size_t ws_size, hipStream_t stream) {
  (void)in_sizes; (void)n_in; (void)out_size; (void)ws_size;
  const float* q  = (const float*)d_in[0];
  const float* k  = (const float*)d_in[1];
  const float* v  = (const float*)d_in[2];
  // d_in[3] is the causal tril mask; applied analytically inside attn_kernel
  const float* wq = (const float*)d_in[4];
  const float* bq = (const float*)d_in[5];
  const float* wk = (const float*)d_in[6];
  const float* bk = (const float*)d_in[7];
  const float* wv = (const float*)d_in[8];
  const float* bv = (const float*)d_in[9];
  const float* wo = (const float*)d_in[10];
  const float* bo = (const float*)d_in[11];
  float* out = (float*)d_out;

  _Float16* ws = (_Float16*)d_ws;
  const size_t elems = (size_t)kM * kD;  // 8M f16 elements per tensor
  _Float16* q16 = ws;
  _Float16* k16 = ws + elems;
  _Float16* v16 = ws + 2 * elems;       // [B,H,HD,S] (transposed for TDM)
  _Float16* a16 = ws + 3 * elems;

  const dim3 ggrid(kM / 128, kD / 64);
  gemm_qkv_kernel<0><<<ggrid, 256, 0, stream>>>(q, wq, bq, q16);
  gemm_qkv_kernel<0><<<ggrid, 256, 0, stream>>>(k, wk, bk, k16);
  gemm_qkv_kernel<1><<<ggrid, 256, 0, stream>>>(v, wv, bv, v16);
  attn_kernel<<<dim3(kS / 64, kB * kH), 128, 0, stream>>>(q16, k16, v16, a16);
  gemm_out_kernel<<<ggrid, 256, 0, stream>>>(a16, wo, bo, out);
}
